// Block_22866405884203
// MI455X (gfx1250) — matrix operand order repaired and checked by reading, not compiled, not run on hardware
//
#include <hip/hip_runtime.h>

// ---------------------------------------------------------------------------
// Transformer block for MI455X (gfx1250, wave32, WMMA f16 -> f32).
// B=8, N=1024, DIM=768, HEADS=12, HEAD_DIM=64, HIDDEN=3072.
// GEMMs: TDM-staged weight tiles + async-to-LDS activation tiles,
// double-buffered, consumed via ds_load_b128 into v_wmma_f32_16x16x32_f16.
// ---------------------------------------------------------------------------

typedef __attribute__((ext_vector_type(16))) _Float16 v16h;
typedef __attribute__((ext_vector_type(8)))  float    v8f;
typedef __attribute__((ext_vector_type(4)))  unsigned u32x4;
typedef __attribute__((ext_vector_type(8)))  unsigned u32x8;

#define DEV __device__ __forceinline__

constexpr int BB   = 8;
constexpr int NN   = 1024;
constexpr int CDIM = 768;
constexpr int NH   = 12;
constexpr int HD   = 64;
constexpr int HID  = 3072;
constexpr int TOK  = BB * NN;           // 8192 tokens

// ---- workspace layout (bytes) ----
constexpr size_t SZ_H16   = (size_t)TOK * CDIM * 2;       // ln1 out, f16
constexpr size_t SZ_WQKV  = (size_t)3 * CDIM * CDIM * 2;  // 2304x768 f16
constexpr size_t SZ_WPROJ = (size_t)CDIM * CDIM * 2;
constexpr size_t SZ_WFC1  = (size_t)HID * CDIM * 2;
constexpr size_t SZ_WFC2  = (size_t)CDIM * HID * 2;
constexpr size_t SZ_Q     = (size_t)TOK * CDIM * 2;       // [B,H,N,64] f16
constexpr size_t SZ_K     = SZ_Q;
constexpr size_t SZ_VT    = SZ_Q;                          // [B,H,64,N] f16
constexpr size_t SZ_O     = SZ_Q;                          // [B,N,768] f16
constexpr size_t SZ_X1    = (size_t)TOK * CDIM * 4;       // residual1, f32
constexpr size_t SZ_H2    = SZ_H16;                        // ln2 out f16
constexpr size_t SZ_F1    = (size_t)TOK * HID * 2;        // gelu(fc1) f16

constexpr size_t OFF_H16   = 0;
constexpr size_t OFF_WQKV  = OFF_H16   + SZ_H16;
constexpr size_t OFF_WPROJ = OFF_WQKV  + SZ_WQKV;
constexpr size_t OFF_WFC1  = OFF_WPROJ + SZ_WPROJ;
constexpr size_t OFF_WFC2  = OFF_WFC1  + SZ_WFC1;
constexpr size_t OFF_Q     = OFF_WFC2  + SZ_WFC2;
constexpr size_t OFF_K     = OFF_Q     + SZ_Q;
constexpr size_t OFF_VT    = OFF_K     + SZ_K;
constexpr size_t OFF_O     = OFF_VT    + SZ_VT;
constexpr size_t OFF_X1    = OFF_O     + SZ_O;
constexpr size_t OFF_H2    = OFF_X1    + SZ_X1;
constexpr size_t OFF_F1    = OFF_H2    + SZ_H2;

// ---------------------------------------------------------------------------
// WMMA + fragment helpers
// ---------------------------------------------------------------------------
DEV v8f vzero8() {
  v8f z;
#pragma unroll
  for (int i = 0; i < 8; ++i) z[i] = 0.0f;
  return z;
}

DEV v8f wmma_f16(v16h a, v16h b, v8f c) {
  return __builtin_amdgcn_wmma_f32_16x16x32_f16(
      /*neg_a=*/false, a, /*neg_b=*/false, b,
      /*c_mod=*/(short)0, c, /*reuse_a=*/false, /*reuse_b=*/false);
}

// A fragment (16x32, f16) from global: lane row = lane&15; half = lane>>4.
DEV v16h load_a_frag(const _Float16* __restrict__ A, int lda, int row0, int k,
                     int lane) {
  int r  = row0 + (lane & 15);
  int h8 = (lane >> 4) << 3;
  const _Float16* p = A + (size_t)r * lda + k;
  union { uint4 u[2]; v16h v; } uu;
  uu.u[0] = *reinterpret_cast<const uint4*>(p + 2 * h8);
  uu.u[1] = *reinterpret_cast<const uint4*>(p + 2 * h8 + 8);
  return uu.v;
}

// B fragment (32x16, f16) from global, B[k,n] = W[col0+n, k]:
DEV v16h load_b_frag(const _Float16* __restrict__ W, int ldw, int col0, int k,
                     int lane) {
  int n   = col0 + (lane & 15);
  int h16 = (lane >> 4) << 4;
  const _Float16* p = W + (size_t)n * ldw + k + h16;
  union { uint4 u[2]; v16h v; } uu;
  uu.u[0] = reinterpret_cast<const uint4*>(p)[0];
  uu.u[1] = reinterpret_cast<const uint4*>(p)[1];
  return uu.v;
}

// Fragments from an LDS tile with 40-half (80 B) row stride
DEV v16h lds_a_frag(const _Float16* base, int lo, int hlf) {
  const _Float16* p = base + (size_t)lo * 40;
  union { uint4 u[2]; v16h v; } uu;
  uu.u[0] = *reinterpret_cast<const uint4*>(p + (hlf << 4));
  uu.u[1] = *reinterpret_cast<const uint4*>(p + (hlf << 4) + 8);
  return uu.v;
}
DEV v16h lds_b_frag(const _Float16* base, int lo, int hlf) {
  const _Float16* p = base + (size_t)lo * 40 + (hlf << 4);
  union { uint4 u[2]; v16h v; } uu;
  uu.u[0] = reinterpret_cast<const uint4*>(p)[0];
  uu.u[1] = reinterpret_cast<const uint4*>(p)[1];
  return uu.v;
}

// ---------------------------------------------------------------------------
// CDNA5 async data movers
// ---------------------------------------------------------------------------
DEV unsigned lds_addr32(const void* p) {  // generic shared ptr -> 32-bit LDS addr
  return (unsigned)(unsigned long long)p;
}

// one 16-byte chunk, tracked by ASYNCcnt
DEV void async_copy_b128(unsigned lds_addr, const _Float16* gptr) {
  asm volatile("global_load_async_to_lds_b128 %0, %1, off"
               :: "v"(lds_addr), "v"(gptr) : "memory");
}

template <int N>
DEV void wait_async() {
#if __has_builtin(__builtin_amdgcn_s_wait_asynccnt)
  __builtin_amdgcn_s_wait_asynccnt(N);
#else
  if (N == 0) asm volatile("s_wait_asynccnt 0x0" ::: "memory");
  else        asm volatile("s_wait_asynccnt 0x1" ::: "memory");
#endif
}

template <int N>
DEV void wait_tensor() {
#if __has_builtin(__builtin_amdgcn_s_wait_tensorcnt)
  __builtin_amdgcn_s_wait_tensorcnt(N);
#else
  if (N == 0) asm volatile("s_wait_tensorcnt 0x0" ::: "memory");
  else        asm volatile("s_wait_tensorcnt 0x1" ::: "memory");
#endif
}

// TDM: load a [256 rows x 32 elem] f16 tile (row stride K elems) into LDS,
// with hardware padding 16 DWORDs -> +4 DWORDs (80-byte LDS row stride).
DEV void issue_b_tdm(const _Float16* W, int K, int nBase, int kk,
                     unsigned lds_base) {
  unsigned long long ga =
      (unsigned long long)(W + (size_t)nBase * K + kk);
  u32x4 g0;
  g0[0] = 1u;                                   // count=1, user-mode
  g0[1] = lds_base;                             // lds_addr
  g0[2] = (unsigned)ga;                         // global_addr[31:0]
  g0[3] = (unsigned)((ga >> 32) & 0x01FFFFFFu)  // global_addr[56:32]
          | (2u << 30);                         // type = 2 ("image")
  u32x8 g1;
  g1[0] = (1u << 16)      // data_size = 2 bytes
        | (1u << 20)      // pad_enable
        | (3u << 22)      // pad_interval: 16 DWORDs (one 64B row)
        | (3u << 25);     // pad_amount: 4 DWORDs (16B)
  g1[1] = ((unsigned)K & 0xFFFFu) << 16;        // tensor_dim0[15:0]
  g1[2] = ((unsigned)K >> 16) & 0xFFFFu;        // tensor_dim0[31:16], dim1 lo=0
  g1[3] = 1u | (32u << 16);                     // tensor_dim1=1<<16, tile_dim0=32
  g1[4] = 256u;                                 // tile_dim1=256, tile_dim2=0
  g1[5] = (unsigned)K;                          // tensor_dim0_stride[31:0]
  g1[6] = 0u;
  g1[7] = 0u;
  asm volatile("tensor_load_to_lds %0, %1" :: "s"(g0), "s"(g1) : "memory");
}

// ---------------------------------------------------------------------------
// GEMM core with LDS double buffering:
// block tile 64(M) x 256(N), 8 waves -> each wave 32x64 (acc[2][4]).
// A tile: 256 async b128 chunks/step.  B tile: one TDM descriptor/step.
// ---------------------------------------------------------------------------
DEV void gemm_core_lds(const _Float16* __restrict__ A,
                       const _Float16* __restrict__ W, int K, int mBlockBase,
                       int nBlockBase, v8f acc[2][4]) {
  __shared__ __align__(16) _Float16 tA[2][64][40];
  __shared__ __align__(16) _Float16 tB[2][256][40];

  const int tid = threadIdx.x;
  const int lane = tid & 31, wave = tid >> 5;
  const int hlf = lane >> 4, lo = lane & 15;
  const int mW = (wave >> 2) * 32;   // wave M offset within block tile
  const int nW = (wave & 3) * 64;    // wave N offset within block tile

  const int aRow = tid >> 2, aCh = tid & 3;  // this thread's A chunk
  unsigned aBase[2] = { lds_addr32(&tA[0][0][0]), lds_addr32(&tA[1][0][0]) };
  unsigned bBase[2] = { lds_addr32(&tB[0][0][0]), lds_addr32(&tB[1][0][0]) };

  // prologue: stage k=0 into buffer 0
  async_copy_b128(aBase[0] + aRow * 80 + aCh * 16,
                  A + (size_t)(mBlockBase + aRow) * K + aCh * 8);
  if (wave == 0) issue_b_tdm(W, K, nBlockBase, 0, bBase[0]);

  int buf = 0;
  for (int kk = 0; kk < K; kk += 32) {
    const bool more = (kk + 32) < K;
    if (more) {
      async_copy_b128(aBase[buf ^ 1] + aRow * 80 + aCh * 16,
                      A + (size_t)(mBlockBase + aRow) * K + kk + 32 + aCh * 8);
      if (wave == 0) issue_b_tdm(W, K, nBlockBase, kk + 32, bBase[buf ^ 1]);
      wait_async<1>();
      if (wave == 0) wait_tensor<1>();
    } else {
      wait_async<0>();
      if (wave == 0) wait_tensor<0>();
    }
    __syncthreads();

    v16h a0 = lds_a_frag(&tA[buf][mW][0],      lo, hlf);
    v16h a1 = lds_a_frag(&tA[buf][mW + 16][0], lo, hlf);
    v16h b0 = lds_b_frag(&tB[buf][nW][0],      lo, hlf);
    v16h b1 = lds_b_frag(&tB[buf][nW + 16][0], lo, hlf);
    v16h b2 = lds_b_frag(&tB[buf][nW + 32][0], lo, hlf);
    v16h b3 = lds_b_frag(&tB[buf][nW + 48][0], lo, hlf);
    acc[0][0] = wmma_f16(a0, b0, acc[0][0]);
    acc[0][1] = wmma_f16(a0, b1, acc[0][1]);
    acc[0][2] = wmma_f16(a0, b2, acc[0][2]);
    acc[0][3] = wmma_f16(a0, b3, acc[0][3]);
    acc[1][0] = wmma_f16(a1, b0, acc[1][0]);
    acc[1][1] = wmma_f16(a1, b1, acc[1][1]);
    acc[1][2] = wmma_f16(a1, b2, acc[1][2]);
    acc[1][3] = wmma_f16(a1, b3, acc[1][3]);

    __syncthreads();
    buf ^= 1;
  }
}

// ---------------------------------------------------------------------------
// Elementwise / LayerNorm kernels
// ---------------------------------------------------------------------------
__global__ __launch_bounds__(256) void cast_f32_to_f16_kernel(
    const float* __restrict__ src, _Float16* __restrict__ dst, int n) {
  for (int i = blockIdx.x * blockDim.x + threadIdx.x; i < n;
       i += gridDim.x * blockDim.x)
    dst[i] = (_Float16)src[i];
}

__global__ __launch_bounds__(256) void ln_kernel(
    const float* __restrict__ x, const float* __restrict__ w,
    const float* __restrict__ b, _Float16* __restrict__ out) {
  __shared__ float red[2][8];
  const int row = blockIdx.x;
  const int t = threadIdx.x;
  const float* xr = x + (size_t)row * CDIM;
  float v0 = xr[t], v1 = xr[t + 256], v2 = xr[t + 512];
  float s = v0 + v1 + v2;
  float q = v0 * v0 + v1 * v1 + v2 * v2;
#pragma unroll
  for (int m = 1; m < 32; m <<= 1) {
    s += __shfl_xor(s, m, 32);
    q += __shfl_xor(q, m, 32);
  }
  if ((t & 31) == 0) { red[0][t >> 5] = s; red[1][t >> 5] = q; }
  __syncthreads();
  float S = 0.f, Q = 0.f;
#pragma unroll
  for (int i = 0; i < 8; ++i) { S += red[0][i]; Q += red[1][i]; }
  const float mean = S * (1.0f / CDIM);
  const float var  = Q * (1.0f / CDIM) - mean * mean;
  const float rstd = rsqrtf(var + 1e-5f);
  _Float16* o = out + (size_t)row * CDIM;
  o[t]       = (_Float16)((v0 - mean) * rstd * w[t]       + b[t]);
  o[t + 256] = (_Float16)((v1 - mean) * rstd * w[t + 256] + b[t + 256]);
  o[t + 512] = (_Float16)((v2 - mean) * rstd * w[t + 512] + b[t + 512]);
}

// ---------------------------------------------------------------------------
// GEMM kernels (block = 256 thr = 8 waves, block tile 64 x 256)
// ---------------------------------------------------------------------------
__global__ __launch_bounds__(256) void gemm_qkv_kernel(
    const _Float16* __restrict__ A, const _Float16* __restrict__ W,
    _Float16* __restrict__ q16, _Float16* __restrict__ k16,
    _Float16* __restrict__ vT16) {
  const int lane = threadIdx.x & 31, wave = threadIdx.x >> 5;
  const int mBase = blockIdx.x * 64 + (wave >> 2) * 32;
  const int nBase = blockIdx.y * 256 + (wave & 3) * 64;
  v8f acc[2][4];
#pragma unroll
  for (int i = 0; i < 2; ++i)
#pragma unroll
    for (int j = 0; j < 4; ++j) acc[i][j] = vzero8();

  gemm_core_lds(A, W, CDIM, blockIdx.x * 64, blockIdx.y * 256, acc);

  const int hlf = lane >> 4, lo = lane & 15;
#pragma unroll
  for (int i = 0; i < 2; ++i)
#pragma unroll
    for (int nt = 0; nt < 4; ++nt)
#pragma unroll
      for (int v = 0; v < 8; ++v) {
        const int m = mBase + i * 16 + v + (hlf << 3);   // token
        const int f = nBase + nt * 16 + lo;              // qkv feature
        const int b = m >> 10, n = m & 1023;
        const int three = f / CDIM;
        const int rem = f - three * CDIM;
        const int head = rem >> 6, d = rem & 63;
        const _Float16 val = (_Float16)acc[i][nt][v];
        if (three == 0)
          q16[(((size_t)(b * NH + head)) * NN + n) * HD + d] = val;
        else if (three == 1)
          k16[(((size_t)(b * NH + head)) * NN + n) * HD + d] = val;
        else
          vT16[(((size_t)(b * NH + head)) * HD + d) * NN + n] = val;
      }
}

__global__ __launch_bounds__(256) void gemm_proj_kernel(
    const _Float16* __restrict__ A, const _Float16* __restrict__ W,
    const float* __restrict__ bias, const float* __restrict__ xres,
    float* __restrict__ x1) {
  const int lane = threadIdx.x & 31, wave = threadIdx.x >> 5;
  const int mBase = blockIdx.x * 64 + (wave >> 2) * 32;
  const int nBase = blockIdx.y * 256 + (wave & 3) * 64;
  v8f acc[2][4];
#pragma unroll
  for (int i = 0; i < 2; ++i)
#pragma unroll
    for (int j = 0; j < 4; ++j) acc[i][j] = vzero8();

  gemm_core_lds(A, W, CDIM, blockIdx.x * 64, blockIdx.y * 256, acc);

  const int hlf = lane >> 4, lo = lane & 15;
#pragma unroll
  for (int i = 0; i < 2; ++i)
#pragma unroll
    for (int nt = 0; nt < 4; ++nt)
#pragma unroll
      for (int v = 0; v < 8; ++v) {
        const int m = mBase + i * 16 + v + (hlf << 3);
        const int c = nBase + nt * 16 + lo;
        const size_t idx = (size_t)m * CDIM + c;
        x1[idx] = acc[i][nt][v] + bias[c] + xres[idx];
      }
}

__global__ __launch_bounds__(256) void gemm_fc1_kernel(
    const _Float16* __restrict__ A, const _Float16* __restrict__ W,
    const float* __restrict__ bias, _Float16* __restrict__ f1) {
  const int lane = threadIdx.x & 31, wave = threadIdx.x >> 5;
  const int mBase = blockIdx.x * 64 + (wave >> 2) * 32;
  const int nBase = blockIdx.y * 256 + (wave & 3) * 64;
  v8f acc[2][4];
#pragma unroll
  for (int i = 0; i < 2; ++i)
#pragma unroll
    for (int j = 0; j < 4; ++j) acc[i][j] = vzero8();

  gemm_core_lds(A, W, CDIM, blockIdx.x * 64, blockIdx.y * 256, acc);

  const int hlf = lane >> 4, lo = lane & 15;
#pragma unroll
  for (int i = 0; i < 2; ++i)
#pragma unroll
    for (int nt = 0; nt < 4; ++nt)
#pragma unroll
      for (int v = 0; v < 8; ++v) {
        const int m = mBase + i * 16 + v + (hlf << 3);
        const int c = nBase + nt * 16 + lo;
        const float t = acc[i][nt][v] + bias[c];
        const float g = 0.5f * t * (1.0f + erff(t * 0.70710678118654752f));
        f1[(size_t)m * HID + c] = (_Float16)g;
      }
}

__global__ __launch_bounds__(256) void gemm_fc2_kernel(
    const _Float16* __restrict__ A, const _Float16* __restrict__ W,
    const float* __restrict__ bias, const float* __restrict__ x1,
    float* __restrict__ out) {
  const int lane = threadIdx.x & 31, wave = threadIdx.x >> 5;
  const int mBase = blockIdx.x * 64 + (wave >> 2) * 32;
  const int nBase = blockIdx.y * 256 + (wave & 3) * 64;
  v8f acc[2][4];
#pragma unroll
  for (int i = 0; i < 2; ++i)
#pragma unroll
    for (int j = 0; j < 4; ++j) acc[i][j] = vzero8();

  gemm_core_lds(A, W, HID, blockIdx.x * 64, blockIdx.y * 256, acc);

  const int hlf = lane >> 4, lo = lane & 15;
#pragma unroll
  for (int i = 0; i < 2; ++i)
#pragma unroll
    for (int nt = 0; nt < 4; ++nt)
#pragma unroll
      for (int v = 0; v < 8; ++v) {
        const int m = mBase + i * 16 + v + (hlf << 3);
        const int c = nBase + nt * 16 + lo;
        const size_t idx = (size_t)m * CDIM + c;
        out[idx] = acc[i][nt][v] + bias[c] + x1[idx];
      }
}

// ---------------------------------------------------------------------------
// Attention: softmax(q k^T * s) + bias, clamp[0,1], @ V.
// One wave owns 16 query rows of one (b,h). Two WMMA passes over keys:
//  pass 1: online softmax stats; pass 2: prob tiles -> LDS transpose -> @ V^T
// ---------------------------------------------------------------------------
__global__ __launch_bounds__(256) void attn_kernel(
    const _Float16* __restrict__ q16, const _Float16* __restrict__ k16,
    const _Float16* __restrict__ vT16, const float* __restrict__ amat,
    _Float16* __restrict__ o16) {
  __shared__ __align__(16) _Float16 lds[8][16][40];  // per-wave 16x32 (+pad)

  const int lane = threadIdx.x & 31, wave = threadIdx.x >> 5;
  const int bh = blockIdx.x >> 3;            // 0..95
  const int rowbase = (blockIdx.x & 7) * 128 + wave * 16;
  const int b = bh / NH, h = bh - b * NH;
  const int hlf = lane >> 4, lo = lane & 15;
  const float scale = 0.125f;                // 1/sqrt(64)

  const _Float16* qp = q16 + (size_t)bh * NN * HD;
  const _Float16* kp = k16 + (size_t)bh * NN * HD;
  const _Float16* vp = vT16 + (size_t)bh * HD * NN;

  const v16h qa0 = load_a_frag(qp, HD, rowbase, 0,  lane);
  const v16h qa1 = load_a_frag(qp, HD, rowbase, 32, lane);

  // ---- pass 1: online row max / sum ----
  float m[8], s[8];
#pragma unroll
  for (int v = 0; v < 8; ++v) { m[v] = -1e30f; s[v] = 0.0f; }

  for (int j = 0; j < NN; j += 16) {
    v16h kb0 = load_b_frag(kp, HD, j, 0,  lane);
    v16h kb1 = load_b_frag(kp, HD, j, 32, lane);
    v8f c = vzero8();
    c = wmma_f16(qa0, kb0, c);
    c = wmma_f16(qa1, kb1, c);
#pragma unroll
    for (int v = 0; v < 8; ++v) {
      float x = c[v] * scale;
      float xm = x;
#pragma unroll
      for (int msk = 1; msk < 16; msk <<= 1)
        xm = fmaxf(xm, __shfl_xor(xm, msk, 32));
      const float nm = fmaxf(m[v], xm);
      float e = __expf(x - nm);
#pragma unroll
      for (int msk = 1; msk < 16; msk <<= 1) e += __shfl_xor(e, msk, 32);
      s[v] = s[v] * __expf(m[v] - nm) + e;
      m[v] = nm;
    }
  }
  float inv[8];
#pragma unroll
  for (int v = 0; v < 8; ++v) inv[v] = 1.0f / s[v];

  // ---- pass 2: p = clip(softmax + bias, 0, 1); O += p @ V ----
  v8f oacc[4];
#pragma unroll
  for (int nt = 0; nt < 4; ++nt) oacc[nt] = vzero8();

  for (int j = 0; j < NN; j += 32) {
    v8f c0 = vzero8(), c1 = vzero8();
    {
      v16h kb0 = load_b_frag(kp, HD, j, 0,  lane);
      v16h kb1 = load_b_frag(kp, HD, j, 32, lane);
      c0 = wmma_f16(qa0, kb0, c0);
      c0 = wmma_f16(qa1, kb1, c0);
      kb0 = load_b_frag(kp, HD, j + 16, 0,  lane);
      kb1 = load_b_frag(kp, HD, j + 16, 32, lane);
      c1 = wmma_f16(qa0, kb0, c1);
      c1 = wmma_f16(qa1, kb1, c1);
    }
#pragma unroll
    for (int v = 0; v < 8; ++v) {
      const int r = v + (hlf << 3);
      const size_t brow = ((size_t)b * NN + rowbase + r) * NN;
      float p0 = __expf(c0[v] * scale - m[v]) * inv[v] + amat[brow + j + lo];
      float p1 = __expf(c1[v] * scale - m[v]) * inv[v] + amat[brow + j + 16 + lo];
      p0 = fminf(fmaxf(p0, 0.0f), 1.0f);
      p1 = fminf(fmaxf(p1, 0.0f), 1.0f);
      lds[wave][r][lo]      = (_Float16)p0;
      lds[wave][r][16 + lo] = (_Float16)p1;
    }
    // wave-private LDS round-trip: DS ops are in-order per wave; wait stores
    asm volatile("s_wait_dscnt 0" ::: "memory");
    v16h af;
    {
      union { uint4 u[2]; v16h v; } uu;
      uu.u[0] = *reinterpret_cast<const uint4*>(&lds[wave][lo][(hlf << 4)]);
      uu.u[1] = *reinterpret_cast<const uint4*>(&lds[wave][lo][(hlf << 4) + 8]);
      af = uu.v;
    }
    asm volatile("" ::: "memory");
#pragma unroll
    for (int nt = 0; nt < 4; ++nt) {
      const _Float16* pv = vp + (size_t)(nt * 16 + lo) * NN + j + (hlf << 4);
      union { uint4 u[2]; v16h v; } ub;
      ub.u[0] = reinterpret_cast<const uint4*>(pv)[0];
      ub.u[1] = reinterpret_cast<const uint4*>(pv)[1];
      oacc[nt] = wmma_f16(af, ub.v, oacc[nt]);
    }
  }

  // store O as [B, N, H*64] f16 (input to projection GEMM)
#pragma unroll
  for (int nt = 0; nt < 4; ++nt)
#pragma unroll
    for (int v = 0; v < 8; ++v) {
      const int r = rowbase + v + (hlf << 3);
      const int c = h * HD + nt * 16 + lo;
      o16[((size_t)b * NN + r) * CDIM + c] = (_Float16)oacc[nt][v];
    }
}

// ---------------------------------------------------------------------------
// Host launcher
// ---------------------------------------------------------------------------
extern "C" void kernel_launch(void* const* d_in, const int* in_sizes, int n_in,
                              void* d_out, int out_size, void* d_ws,
                              size_t ws_size, hipStream_t stream) {
  (void)in_sizes; (void)n_in; (void)out_size; (void)ws_size;
  const float* x      = (const float*)d_in[0];
  const float* amat   = (const float*)d_in[1];
  const float* w_qkv  = (const float*)d_in[2];
  const float* w_proj = (const float*)d_in[3];
  const float* b_proj = (const float*)d_in[4];
  const float* ln1w   = (const float*)d_in[5];
  const float* ln1b   = (const float*)d_in[6];
  const float* ln2w   = (const float*)d_in[7];
  const float* ln2b   = (const float*)d_in[8];
  const float* w_fc1  = (const float*)d_in[9];
  const float* b_fc1  = (const float*)d_in[10];
  const float* w_fc2  = (const float*)d_in[11];
  const float* b_fc2  = (const float*)d_in[12];
  float* out = (float*)d_out;

  char* ws = (char*)d_ws;
  _Float16* h16    = (_Float16*)(ws + OFF_H16);
  _Float16* wqkv16 = (_Float16*)(ws + OFF_WQKV);
  _Float16* wproj16= (_Float16*)(ws + OFF_WPROJ);
  _Float16* wfc116 = (_Float16*)(ws + OFF_WFC1);
  _Float16* wfc216 = (_Float16*)(ws + OFF_WFC2);
  _Float16* q16    = (_Float16*)(ws + OFF_Q);
  _Float16* k16    = (_Float16*)(ws + OFF_K);
  _Float16* vT16   = (_Float16*)(ws + OFF_VT);
  _Float16* o16    = (_Float16*)(ws + OFF_O);
  float*    x1     = (float*)   (ws + OFF_X1);
  _Float16* h2     = (_Float16*)(ws + OFF_H2);
  _Float16* f1     = (_Float16*)(ws + OFF_F1);

  // 1) cast weights to f16 (layout preserved: [out_features, in_features])
  cast_f32_to_f16_kernel<<<2048, 256, 0, stream>>>(w_qkv,  wqkv16, 3 * CDIM * CDIM);
  cast_f32_to_f16_kernel<<<1024, 256, 0, stream>>>(w_proj, wproj16, CDIM * CDIM);
  cast_f32_to_f16_kernel<<<2048, 256, 0, stream>>>(w_fc1,  wfc116, HID * CDIM);
  cast_f32_to_f16_kernel<<<2048, 256, 0, stream>>>(w_fc2,  wfc216, CDIM * HID);

  // 2) LN1
  ln_kernel<<<TOK, 256, 0, stream>>>(x, ln1w, ln1b, h16);

  // 3) QKV GEMM  (M=8192, N=2304, K=768)
  gemm_qkv_kernel<<<dim3(TOK / 64, (3 * CDIM) / 256), 256, 0, stream>>>(
      h16, wqkv16, q16, k16, vT16);

  // 4) attention (96 bh-pairs x 8 row-blocks)
  attn_kernel<<<BB * NH * 8, 256, 0, stream>>>(q16, k16, vT16, amat, o16);

  // 5) projection + bias + residual -> x1
  gemm_proj_kernel<<<dim3(TOK / 64, CDIM / 256), 256, 0, stream>>>(
      o16, wproj16, b_proj, x, x1);

  // 6) LN2
  ln_kernel<<<TOK, 256, 0, stream>>>(x1, ln2w, ln2b, h2);

  // 7) fc1 + bias + exact GELU
  gemm_fc1_kernel<<<dim3(TOK / 64, HID / 256), 256, 0, stream>>>(
      h2, wfc116, b_fc1, f1);

  // 8) fc2 + bias + residual -> out
  gemm_fc2_kernel<<<dim3(TOK / 64, CDIM / 256), 256, 0, stream>>>(
      f1, wfc216, b_fc2, x1, out);
}
